// SpatialMemoryNet_62758062129655
// MI455X (gfx1250) — matrix operand order repaired and checked by reading, not compiled, not run on hardware
//
#include <hip/hip_runtime.h>

typedef __attribute__((ext_vector_type(16))) _Float16 v16h;
typedef __attribute__((ext_vector_type(8)))  _Float16 v8h;
typedef __attribute__((ext_vector_type(8)))  float    v8f;

#define B_TOT 131072
#define T_    15
#define WAVES 8
#define ROWS_PER_WG (WAVES * 16)

// ---------------- LDS layout (units: _Float16 elements) ----------------
#define OFF_EW1 0                       // enc_w1  Kpad32 x 64  ->  2048
#define OFF_EW2 2048                    // enc_w2  64 x 64      ->  4096
#define OFF_WIH 6144                    // w_ih    64 x 512     -> 32768
#define OFF_WHH 38912                   // w_hh    128 x 512    -> 65536
#define OFF_STG 104448                  // 8 waves * (16x128 h + 16x64 z)
#define SMEM_HALFS (OFF_STG + WAVES * 3072)   // 129024 halfs = 258048 bytes
// Head weights reuse the w_ih/w_hh region after a barrier:
#define OFF_CW1 6144
#define OFF_CW2 (OFF_CW1 + 16384)
#define OFF_LW1 (OFF_CW2 + 4096)
#define OFF_LW2 (OFF_LW1 + 8192)

#if __has_builtin(__builtin_amdgcn_tanhf)
#define TANHF(x) __builtin_amdgcn_tanhf(x)
#else
#define TANHF(x) (1.f - 2.f * __builtin_amdgcn_rcpf(1.f + __expf(2.f * (x))))
#endif

// sigmoid(x + b) with hb = 0.5*b precomputed: bias folds into the scale FMA.
__device__ __forceinline__ float fsig_b(float x, float hb) {
  return __builtin_fmaf(0.5f, TANHF(__builtin_fmaf(0.5f, x, hb)), 0.5f);
}

__device__ __forceinline__ v8f zero8() {
  v8f z;
#pragma unroll
  for (int i = 0; i < 8; ++i) z[i] = 0.f;
  return z;  // folds to the WMMA inline-0 C operand
}

__device__ __forceinline__ v8f wmma_(v16h a, v16h b, v8f c) {
  // D = A(16x32 f16) * B(32x16 f16) + C(16x16 f32)
  return __builtin_amdgcn_wmma_f32_16x16x32_f16(false, a, false, b, (short)0, c,
                                                false, false);
}

// Cooperative f32 -> f16 weight staging into fragment-major LDS.
// B-fragment lane mapping (16-bit B, 32x16 per fragment):
//   lane L: column n = nt*16 + (L&15); element e: K = kc*32 + 8*(L>>4) + (e&7) + 16*(e>>3)
__device__ __forceinline__ void stage_w(const float* __restrict__ W, _Float16* dst,
                                        int Kreal, int Nreal, int Kpad, int Npad,
                                        int tid) {
  const int NT = Npad >> 4;
  const int total = ((Kpad >> 5) * NT) << 9;
  for (int d = tid; d < total; d += 256) {
    int f = d >> 9, r = d & 511, L = r >> 4, e = r & 15;
    int kc = f / NT, nt = f - kc * NT;
    int k = (kc << 5) + ((L >> 4) << 3) + (e & 7) + ((e >> 3) << 4);
    int n = (nt << 4) + (L & 15);
    float v = (k < Kreal && n < Nreal) ? W[k * Nreal + n] : 0.f;
    dst[d] = (_Float16)v;
  }
}

// Load one B fragment: one contiguous 32B read per lane.
__device__ __forceinline__ v16h bfrag(const _Float16* w, int NT, int kc, int nt,
                                      int lane) {
  return *(const v16h*)(w + (((kc * NT + nt) << 9) + (lane << 4)));
}

// Build an A fragment (16x32 f16, ISA interleaved layout) from a plain
// row-major [16][rowHalfs] f16 staging tile. Two contiguous 16B reads/lane.
__device__ __forceinline__ v16h afrag(const _Float16* stg, int rowHalfs, int kb,
                                      int lane) {
  int m = lane & 15, hi = (lane >> 4) & 1;
  const _Float16* p = stg + m * rowHalfs + kb + hi * 8;
  v8h lo = *(const v8h*)p;          // K = kb + 8*hi + 0..7
  v8h up = *(const v8h*)(p + 16);   // K = kb + 8*hi + 16..23
  return __builtin_shufflevector(lo, up, 0, 1, 2, 3, 4, 5, 6, 7, 8, 9, 10, 11,
                                 12, 13, 14, 15);
}

// Scatter a C/D fragment (f32) into a plain [16][rowHalfs] f16 staging tile.
__device__ __forceinline__ void dstore(_Float16* stg, int rowHalfs, int nt,
                                       int lane, v8f d) {
  int ln = lane & 15, hi = (lane >> 4) & 1;
  _Float16* p = stg + (hi * 8) * rowHalfs + nt * 16 + ln;
#pragma unroll
  for (int r = 0; r < 8; ++r) p[r * rowHalfs] = (_Float16)d[r];
}

__device__ __forceinline__ void load_x(float* xv, const float* __restrict__ obs_l,
                                       const float* __restrict__ obs_c,
                                       const float* __restrict__ obs_m,
                                       long bidx) {
  float2 l2 = *(const float2*)(obs_l + bidx * 2);
  float cc = obs_c[bidx];
  float4 m4 = *(const float4*)(obs_m + bidx * 4);
  xv[0] = l2.x; xv[1] = l2.y; xv[2] = cc;
  xv[3] = m4.x; xv[4] = m4.y; xv[5] = m4.z; xv[6] = m4.w;
}

__global__ __launch_bounds__(256) void smn_kernel(
    const float* __restrict__ obs_l, const float* __restrict__ obs_c,
    const float* __restrict__ obs_m, const float* __restrict__ enc_w1,
    const float* __restrict__ enc_b1, const float* __restrict__ enc_w2,
    const float* __restrict__ enc_b2, const float* __restrict__ w_ih,
    const float* __restrict__ w_hh, const float* __restrict__ b_ih,
    const float* __restrict__ b_hh, const float* __restrict__ coord_w1,
    const float* __restrict__ coord_b1, const float* __restrict__ coord_w2,
    const float* __restrict__ coord_b2, const float* __restrict__ label_w1,
    const float* __restrict__ label_b1, const float* __restrict__ label_w2,
    const float* __restrict__ label_b2, float* __restrict__ out_coords,
    float* __restrict__ out_labels) {
  extern __shared__ __attribute__((aligned(32))) _Float16 sm[];

  const int tid = threadIdx.x;
  const int wave = tid >> 5, lane = tid & 31;
  const int ln = lane & 15, hi = lane >> 4;
  const int row0 = blockIdx.x * ROWS_PER_WG + wave * 16;

  // ---- stage all recurrent-path weights to LDS (f16, fragment-major) ----
  stage_w(enc_w1, sm + OFF_EW1, 7, 64, 32, 64, tid);
  stage_w(enc_w2, sm + OFF_EW2, 64, 64, 64, 64, tid);
  stage_w(w_ih, sm + OFF_WIH, 64, 512, 64, 512, tid);
  stage_w(w_hh, sm + OFF_WHH, 128, 512, 128, 512, tid);
  __syncthreads();

  _Float16* stgH = sm + OFF_STG + wave * 3072;  // [16][128] h staging
  _Float16* stgZ = stgH + 2048;                 // [16][64]  z staging

  // ---- preload biases (per-lane column values); i/f/o pre-scaled by 0.5 ----
  float hbi[8], hbf[8], bg_[8], hbo[8];
#pragma unroll
  for (int j = 0; j < 8; ++j) {
    int n = j * 16 + ln;
    hbi[j] = 0.5f * (b_ih[n] + b_hh[n]);
    hbf[j] = 0.5f * (b_ih[128 + n] + b_hh[128 + n]);
    bg_[j] = b_ih[256 + n] + b_hh[256 + n];
    hbo[j] = 0.5f * (b_ih[384 + n] + b_hh[384 + n]);
  }
  float eb1[4], eb2[4];
#pragma unroll
  for (int nt = 0; nt < 4; ++nt) {
    eb1[nt] = enc_b1[nt * 16 + ln];
    eb2[nt] = enc_b2[nt * 16 + ln];
  }

  // ---- recurrent state: c in f32 C/D layout, h as f16 A fragments ----
  v8f cst[8];
#pragma unroll
  for (int j = 0; j < 8; ++j) cst[j] = zero8();
  v16h hA[4];
#pragma unroll
  for (int kc = 0; kc < 4; ++kc) {
#pragma unroll
    for (int e = 0; e < 16; ++e) hA[kc][e] = (_Float16)0.f;
  }

  // ---- observation double-buffer (7 features/row; hi==1 lanes stay 0) ----
  const long rowbase = (long)(row0 + ln) * T_;
  float xv[7];
#pragma unroll
  for (int k = 0; k < 7; ++k) xv[k] = 0.f;
  if (hi == 0) load_x(xv, obs_l, obs_c, obs_m, rowbase);

#pragma unroll 1
  for (int t = 0; t < T_; ++t) {
    // prefetch next step's observations early (hidden behind this step)
    float xn[7];
#pragma unroll
    for (int k = 0; k < 7; ++k) xn[k] = 0.f;
    int tn = (t + 1 < T_) ? t + 1 : t;
    if (hi == 0) load_x(xn, obs_l, obs_c, obs_m, rowbase + tn);

    // ---- build x_t A-fragment (K padded 7 -> 32 with zeros) ----
    v16h ax;
#pragma unroll
    for (int e = 0; e < 16; ++e) ax[e] = (_Float16)0.f;
#pragma unroll
    for (int e = 0; e < 7; ++e) ax[e] = (_Float16)xv[e];

    // ---- encoder layer 1: 16x7 @ 7x64 ----
#pragma unroll
    for (int nt = 0; nt < 4; ++nt) {
      v8f acc = wmma_(ax, bfrag(sm + OFF_EW1, 4, 0, nt, lane), zero8());
#pragma unroll
      for (int r = 0; r < 8; ++r) acc[r] = fmaxf(acc[r] + eb1[nt], 0.f);
      dstore(stgZ, 64, nt, lane, acc);
    }
    v16h zA0 = afrag(stgZ, 64, 0, lane);
    v16h zA1 = afrag(stgZ, 64, 32, lane);

    // ---- encoder layer 2: 16x64 @ 64x64 ----
    v8f z2[4];
#pragma unroll
    for (int nt = 0; nt < 4; ++nt) {
      v8f acc = wmma_(zA0, bfrag(sm + OFF_EW2, 4, 0, nt, lane), zero8());
      acc = wmma_(zA1, bfrag(sm + OFF_EW2, 4, 1, nt, lane), acc);
#pragma unroll
      for (int r = 0; r < 8; ++r) acc[r] = fmaxf(acc[r] + eb2[nt], 0.f);
      z2[nt] = acc;
    }
#pragma unroll
    for (int nt = 0; nt < 4; ++nt) dstore(stgZ, 64, nt, lane, z2[nt]);
    zA0 = afrag(stgZ, 64, 0, lane);
    zA1 = afrag(stgZ, 64, 32, lane);

    // ---- LSTM gates: z@w_ih + h@w_hh (C starts at inline 0) ----
#pragma unroll
    for (int j = 0; j < 8; ++j) {
      v8f gi = zero8(), gf = zero8(), gg = zero8(), go = zero8();
#pragma unroll
      for (int kc = 0; kc < 2; ++kc) {
        v16h a = (kc == 0) ? zA0 : zA1;
        gi = wmma_(a, bfrag(sm + OFF_WIH, 32, kc, j, lane), gi);
        gf = wmma_(a, bfrag(sm + OFF_WIH, 32, kc, j + 8, lane), gf);
        gg = wmma_(a, bfrag(sm + OFF_WIH, 32, kc, j + 16, lane), gg);
        go = wmma_(a, bfrag(sm + OFF_WIH, 32, kc, j + 24, lane), go);
      }
#pragma unroll
      for (int kc = 0; kc < 4; ++kc) {
        gi = wmma_(hA[kc], bfrag(sm + OFF_WHH, 32, kc, j, lane), gi);
        gf = wmma_(hA[kc], bfrag(sm + OFF_WHH, 32, kc, j + 8, lane), gf);
        gg = wmma_(hA[kc], bfrag(sm + OFF_WHH, 32, kc, j + 16, lane), gg);
        go = wmma_(hA[kc], bfrag(sm + OFF_WHH, 32, kc, j + 24, lane), go);
      }
#pragma unroll
      for (int r = 0; r < 8; ++r) {
        float iv = fsig_b(gi[r], hbi[j]);
        float fv = fsig_b(gf[r], hbf[j]);
        float gv = TANHF(gg[r] + bg_[j]);
        float ov = fsig_b(go[r], hbo[j]);
        float c = fv * cst[j][r] + iv * gv;
        cst[j][r] = c;
        float h = ov * TANHF(c);
        stgH[(r + 8 * hi) * 128 + j * 16 + ln] = (_Float16)h;
      }
    }
#pragma unroll
    for (int kc = 0; kc < 4; ++kc) hA[kc] = afrag(stgH, 128, kc * 32, lane);

#pragma unroll
    for (int k = 0; k < 7; ++k) xv[k] = xn[k];
  }

  // ---- restage head weights over the (now dead) LSTM weight region ----
  __syncthreads();
  stage_w(coord_w1, sm + OFF_CW1, 128, 128, 128, 128, tid);
  stage_w(coord_w2, sm + OFF_CW2, 128, 30, 128, 32, tid);
  stage_w(label_w1, sm + OFF_LW1, 128, 64, 128, 64, tid);
  stage_w(label_w2, sm + OFF_LW2, 64, 15, 64, 16, tid);
  __syncthreads();

  // ---- coord head: relu(h@W1+b1) @ W2 + b2 ----
#pragma unroll
  for (int nt = 0; nt < 8; ++nt) {
    v8f acc = zero8();
#pragma unroll
    for (int kc = 0; kc < 4; ++kc)
      acc = wmma_(hA[kc], bfrag(sm + OFF_CW1, 8, kc, nt, lane), acc);
    float b = coord_b1[nt * 16 + ln];
#pragma unroll
    for (int r = 0; r < 8; ++r) acc[r] = fmaxf(acc[r] + b, 0.f);
    dstore(stgH, 128, nt, lane, acc);
  }
  v16h c1A[4];
#pragma unroll
  for (int kc = 0; kc < 4; ++kc) c1A[kc] = afrag(stgH, 128, kc * 32, lane);
#pragma unroll
  for (int nt = 0; nt < 2; ++nt) {
    int n = nt * 16 + ln;
    v8f acc = zero8();
#pragma unroll
    for (int kc = 0; kc < 4; ++kc)
      acc = wmma_(c1A[kc], bfrag(sm + OFF_CW2, 2, kc, nt, lane), acc);
    if (n < 30) {
      float b = coord_b2[n];
#pragma unroll
      for (int r = 0; r < 8; ++r)
        out_coords[(long)(row0 + r + 8 * hi) * 30 + n] = acc[r] + b;
    }
  }

  // ---- label head: relu(h@W1+b1) @ W2 + b2 ----
#pragma unroll
  for (int nt = 0; nt < 4; ++nt) {
    v8f acc = zero8();
#pragma unroll
    for (int kc = 0; kc < 4; ++kc)
      acc = wmma_(hA[kc], bfrag(sm + OFF_LW1, 4, kc, nt, lane), acc);
    float b = label_b1[nt * 16 + ln];
#pragma unroll
    for (int r = 0; r < 8; ++r) acc[r] = fmaxf(acc[r] + b, 0.f);
    dstore(stgZ, 64, nt, lane, acc);
  }
  v16h l1A0 = afrag(stgZ, 64, 0, lane);
  v16h l1A1 = afrag(stgZ, 64, 32, lane);
  {
    int n = ln;
    v8f acc = zero8();
    acc = wmma_(l1A0, bfrag(sm + OFF_LW2, 1, 0, 0, lane), acc);
    acc = wmma_(l1A1, bfrag(sm + OFF_LW2, 1, 1, 0, lane), acc);
    if (n < 15) {
      float b = label_b2[n];
#pragma unroll
      for (int r = 0; r < 8; ++r)
        out_labels[(long)(row0 + r + 8 * hi) * 15 + n] = acc[r] + b;
    }
  }
}

extern "C" void kernel_launch(void* const* d_in, const int* in_sizes, int n_in,
                              void* d_out, int out_size, void* d_ws,
                              size_t ws_size, hipStream_t stream) {
  (void)in_sizes; (void)n_in; (void)out_size; (void)d_ws; (void)ws_size;
  const float* obs_l = (const float*)d_in[0];
  const float* obs_c = (const float*)d_in[1];
  const float* obs_m = (const float*)d_in[2];
  const float* enc_w1 = (const float*)d_in[3];
  const float* enc_b1 = (const float*)d_in[4];
  const float* enc_w2 = (const float*)d_in[5];
  const float* enc_b2 = (const float*)d_in[6];
  const float* w_ih = (const float*)d_in[7];
  const float* w_hh = (const float*)d_in[8];
  const float* b_ih = (const float*)d_in[9];
  const float* b_hh = (const float*)d_in[10];
  const float* coord_w1 = (const float*)d_in[11];
  const float* coord_b1 = (const float*)d_in[12];
  const float* coord_w2 = (const float*)d_in[13];
  const float* coord_b2 = (const float*)d_in[14];
  const float* label_w1 = (const float*)d_in[15];
  const float* label_b1 = (const float*)d_in[16];
  const float* label_w2 = (const float*)d_in[17];
  const float* label_b2 = (const float*)d_in[18];
  float* out_coords = (float*)d_out;
  float* out_labels = out_coords + (size_t)B_TOT * 30;

  const size_t shbytes = (size_t)SMEM_HALFS * 2;  // 258048 B (< 320KB/WG)
  hipFuncSetAttribute((const void*)smn_kernel,
                      hipFuncAttributeMaxDynamicSharedMemorySize, (int)shbytes);
  dim3 grid(B_TOT / ROWS_PER_WG), block(WAVES * 32);
  smn_kernel<<<grid, block, shbytes, stream>>>(
      obs_l, obs_c, obs_m, enc_w1, enc_b1, enc_w2, enc_b2, w_ih, w_hh, b_ih,
      b_hh, coord_w1, coord_b1, coord_w2, coord_b2, label_w1, label_b1,
      label_w2, label_b2, out_coords, out_labels);
}
